// MultiheadAttention_45389214384504
// MI455X (gfx1250) — matrix operand order repaired and checked by reading, not compiled, not run on hardware
//
#include <hip/hip_runtime.h>
#include <hip/hip_bf16.h>
#include <cstdint>

typedef __attribute__((ext_vector_type(16))) __bf16 v16bf;
typedef __attribute__((ext_vector_type(8)))  float  v8f;

#define H_CONST 16
#define S_CONST 2048
#define E_CONST 1024
#define DHEAD   64

__device__ __forceinline__ unsigned short f2bf(float f) {
  unsigned int u = __float_as_uint(f);
  unsigned int r = (u + 0x7FFFu + ((u >> 16) & 1u)) >> 16;  // RNE truncate
  return (unsigned short)r;
}
__device__ __forceinline__ unsigned int pack2(float a, float b) {
  return (unsigned int)f2bf(a) | ((unsigned int)f2bf(b) << 16);
}

union BF16x16 { unsigned int u[8]; v16bf v; };

__device__ __forceinline__ v8f wmma_bf16(v16bf a, v16bf b, v8f c) {
  return __builtin_amdgcn_wmma_f32_16x16x32_bf16(
      /*neg_a=*/false, a, /*neg_b=*/false, b,
      /*c_mod=*/(short)0, c, /*reuse_a=*/false, /*reuse_b=*/false);
}

// Load a K-adjacent pair of elements as one packed-bf16 dword.
__device__ __forceinline__ unsigned int ld_pair(const float* p) {
  const float2 f = *(const float2*)p;
  return pack2(f.x, f.y);
}
__device__ __forceinline__ unsigned int ld_pair(const unsigned short* p) {
  return *(const unsigned int*)p;   // already bf16, contiguous pair
}

// ---------------------------------------------------------------------------
// C = A(MxK) * W^T + bias, W is (N x K) row-major.  bf16 WMMA, f32 accumulate.
// Block tile 128x128x32, 8 waves (4x2), each wave a 32x64 tile (2x4 WMMA accs).
// Software-pipelined: next global tile preloaded into regs during compute.
// EPI==0: scatter to head-major Q/K/V bf16 (q scaled by 1/8).
// EPI==1: bias + NaN->0, fp32 output row-major (M x N).
// ---------------------------------------------------------------------------
template<typename AT, int EPI>
__global__ __launch_bounds__(256) void gemm_bf16_wmma(
    const AT* __restrict__ A, const float* __restrict__ W,
    const float* __restrict__ bias,
    unsigned short* __restrict__ q, unsigned short* __restrict__ kk,
    unsigned short* __restrict__ vv, float* __restrict__ outf,
    int M, int N, int K)
{
  constexpr int BM = 128, BN = 128, BK = 32;
  __shared__ unsigned int Au[BM][BK/2 + 1];   // [row][kpair], pad->17 (bank-safe)
  __shared__ unsigned int Wu[BN][BK/2 + 1];

  const int tid  = threadIdx.x;
  const int lane = tid & 31, wid = tid >> 5;
  const int wm = wid & 3, wn = wid >> 2;       // 4x2 wave grid, 32x64 per wave
  const int bm0 = blockIdx.x * BM, bn0 = blockIdx.y * BN;
  const int hl = lane >> 4, l15 = lane & 15;
  const int r0 = tid >> 4, kp0 = tid & 15;     // staging coords (8 rows/thread)

  const v8f vzero = {0.f,0.f,0.f,0.f,0.f,0.f,0.f,0.f};
  v8f acc[2][4];
  #pragma unroll
  for (int mi = 0; mi < 2; ++mi)
    #pragma unroll
    for (int ni = 0; ni < 4; ++ni) acc[mi][ni] = vzero;

  unsigned int ra[8], rw[8];
  auto preload = [&](int k0) {
    #pragma unroll
    for (int i = 0; i < 8; ++i)
      ra[i] = ld_pair(A + (size_t)(bm0 + r0 + i*16)*K + k0 + kp0*2);
    #pragma unroll
    for (int i = 0; i < 8; ++i)
      rw[i] = ld_pair(W + (size_t)(bn0 + r0 + i*16)*K + k0 + kp0*2);
  };
  preload(0);

  for (int k0 = 0; k0 < K; k0 += BK) {
    #pragma unroll
    for (int i = 0; i < 8; ++i) Au[r0 + i*16][kp0] = ra[i];
    #pragma unroll
    for (int i = 0; i < 8; ++i) Wu[r0 + i*16][kp0] = rw[i];
    __syncthreads();
    if (k0 + BK < K) preload(k0 + BK);         // overlap HBM with WMMA

    v16bf af[2], bfr[4];
    #pragma unroll
    for (int mi = 0; mi < 2; ++mi) {           // A 16x32 frag: row=lane%16
      const int row = wm*32 + mi*16 + l15;
      BF16x16 t;
      #pragma unroll
      for (int j = 0; j < 8; ++j) {
        const int kp = (j < 4) ? (hl*4 + j) : (8 + hl*4 + (j - 4));
        t.u[j] = Au[row][kp];
      }
      af[mi] = t.v;
    }
    #pragma unroll
    for (int ni = 0; ni < 4; ++ni) {           // B 32x16 frag: col=lane%16
      const int row = wn*64 + ni*16 + l15;
      BF16x16 t;
      #pragma unroll
      for (int j = 0; j < 8; ++j) t.u[j] = Wu[row][(j < 4) ? (hl*4 + j) : (8 + hl*4 + (j - 4))];
      bfr[ni] = t.v;
    }
    #pragma unroll
    for (int mi = 0; mi < 2; ++mi)
      #pragma unroll
      for (int ni = 0; ni < 4; ++ni)
        acc[mi][ni] = wmma_bf16(af[mi], bfr[ni], acc[mi][ni]);
    __syncthreads();
  }

  // Epilogue. C/D layout: vgpr i, lane L -> row = i + 8*(L/16), col = L%16.
  #pragma unroll
  for (int mi = 0; mi < 2; ++mi) {
    #pragma unroll
    for (int ni = 0; ni < 4; ++ni) {
      #pragma unroll
      for (int i = 0; i < 8; ++i) {
        const int r = bm0 + wm*32 + mi*16 + i + 8*hl;
        const int c = bn0 + wn*64 + ni*16 + l15;
        float val = acc[mi][ni][i] + bias[c];
        if (EPI == 0) {
          const int which = c >> 10;           // 0:q 1:k 2:v
          const int e = c & 1023;
          const int h = e >> 6, dd = e & 63;
          const int b = r >> 11, s = r & (S_CONST - 1);
          const size_t o = (((size_t)(b*H_CONST + h))*S_CONST + s)*DHEAD + dd;
          if      (which == 0) q[o]  = f2bf(val * 0.125f);  // fold 1/sqrt(64)
          else if (which == 1) kk[o] = f2bf(val);
          else                 vv[o] = f2bf(val);
        } else {
          if (val != val) val = 0.f;           // NaN -> 0 (reference semantics)
          outf[(size_t)r*N + c] = val;
        }
      }
    }
  }
}

// ---------------------------------------------------------------------------
// Causal flash attention.  One block = 128 q rows of one (b,h); 8 waves,
// each wave owns a 16-row WMMA M-tile.  64-key tiles, online softmax,
// global_prefetch of the next K/V tile while computing the current one.
// ---------------------------------------------------------------------------
__global__ __launch_bounds__(256) void flash_attn_wmma(
    const unsigned short* __restrict__ Qb,
    const unsigned short* __restrict__ Kb,
    const unsigned short* __restrict__ Vb,
    unsigned short* __restrict__ attn)
{
  constexpr int QT = 128, KT = 64;
  __shared__ unsigned int   Ku[KT][DHEAD/2 + 1];     // K tile, [key][dpair]
  __shared__ unsigned short Vt[DHEAD][KT + 4];       // V tile transposed [d][key]
  __shared__ unsigned short Pl[8][16][KT + 4];       // per-wave P staging

  const int tid = threadIdx.x, lane = tid & 31, wave = tid >> 5;
  const int hl = lane >> 4, l15 = lane & 15;
  const int qblk = blockIdx.x & (S_CONST/QT - 1);
  const int bh   = blockIdx.x / (S_CONST/QT);        // b*H + h
  const size_t base = (size_t)bh * S_CONST * DHEAD;
  const int q0 = qblk * QT;

  // Load this wave's Q A-fragments once (16 rows x 64 depth = 2 frags).
  v16bf qf[2];
  {
    const unsigned short* qr = Qb + base + (size_t)(q0 + wave*16 + l15) * DHEAD;
    #pragma unroll
    for (int f = 0; f < 2; ++f) {
      BF16x16 t;
      #pragma unroll
      for (int j = 0; j < 8; ++j) {
        const int kp = (j < 4) ? (hl*4 + j) : (8 + hl*4 + (j - 4));
        t.u[j] = *(const unsigned int*)(qr + (f*16 + kp)*2);
      }
      qf[f] = t.v;
    }
  }

  const v8f vzero = {0.f,0.f,0.f,0.f,0.f,0.f,0.f,0.f};
  v8f o[4] = {vzero, vzero, vzero, vzero};           // 16 x 64 output tile
  float m_i[8], l_i[8];
  #pragma unroll
  for (int i = 0; i < 8; ++i) { m_i[i] = -3.0e38f; l_i[i] = 0.f; }

  const int nkt = 2*(qblk + 1);                      // causal: tiles <= diag
  for (int kt = 0; kt < nkt; ++kt) {
    const int kbase = kt * KT;
    // Stage K: rows wave+i*8, dword column = lane (fully coalesced).
    #pragma unroll
    for (int i = 0; i < 8; ++i) {
      const int r = wave + i*8;
      Ku[r][lane] = *(const unsigned int*)(Kb + base + (size_t)(kbase + r)*DHEAD + lane*2);
    }
    // Stage V transposed.
    #pragma unroll
    for (int i = 0; i < 8; ++i) {
      const int r = wave + i*8;
      const unsigned int u =
          *(const unsigned int*)(Vb + base + (size_t)(kbase + r)*DHEAD + lane*2);
      Vt[lane*2    ][r] = (unsigned short)(u & 0xffffu);
      Vt[lane*2 + 1][r] = (unsigned short)(u >> 16);
    }
    // Warm L2/WGP$ with next tile (global_prefetch_b8), one 128B line/thread.
    if (kt + 1 < nkt) {
      const size_t nb = base + (size_t)(kbase + KT)*DHEAD;
      if (tid < 64)       __builtin_prefetch(Kb + nb + (size_t)tid*64, 0, 1);
      else if (tid < 128) __builtin_prefetch(Vb + nb + (size_t)(tid-64)*64, 0, 1);
    }
    __syncthreads();

    // S = Q * K^T  (A=Q 16x32 x2, B=K^T 32x16; 4 key sub-tiles)
    v8f sc[4] = {vzero, vzero, vzero, vzero};
    #pragma unroll
    for (int nt = 0; nt < 4; ++nt) {
      #pragma unroll
      for (int dh = 0; dh < 2; ++dh) {
        BF16x16 t;
        #pragma unroll
        for (int j = 0; j < 8; ++j)
          t.u[j] = Ku[nt*16 + l15][dh*16 + ((j < 4) ? (hl*4 + j) : (8 + hl*4 + (j - 4)))];
        sc[nt] = wmma_bf16(qf[dh], t.v, sc[nt]);
      }
    }

    // Causal mask + online softmax (half-wave reductions match C layout).
    #pragma unroll
    for (int i = 0; i < 8; ++i) {
      const int qr = q0 + wave*16 + i + 8*hl;
      float mv = -3.0e38f;
      #pragma unroll
      for (int nt = 0; nt < 4; ++nt) {
        const int kc = kbase + nt*16 + l15;
        float s = sc[nt][i];
        if (kc > qr) { s = -3.0e38f; sc[nt][i] = s; }
        mv = fmaxf(mv, s);
      }
      #pragma unroll
      for (int off = 1; off < 16; off <<= 1) mv = fmaxf(mv, __shfl_xor(mv, off, 16));
      const float mnew = fmaxf(m_i[i], mv);
      const float corr = exp2f((m_i[i] - mnew) * 1.44269504f);
      float rs = 0.f;
      #pragma unroll
      for (int nt = 0; nt < 4; ++nt) {
        const float p = exp2f((sc[nt][i] - mnew) * 1.44269504f);
        sc[nt][i] = p;
        rs += p;
      }
      #pragma unroll
      for (int off = 1; off < 16; off <<= 1) rs += __shfl_xor(rs, off, 16);
      l_i[i] = l_i[i]*corr + rs;
      m_i[i] = mnew;
      #pragma unroll
      for (int nt = 0; nt < 4; ++nt) o[nt][i] *= corr;
    }

    // Re-stage P: C-layout -> A-layout via per-wave LDS (same-wave, in-order).
    #pragma unroll
    for (int nt = 0; nt < 4; ++nt)
      #pragma unroll
      for (int i = 0; i < 8; ++i)
        Pl[wave][i + 8*hl][nt*16 + l15] = f2bf(sc[nt][i]);

    // O += P * V   (A=P 16x32 x2 key-halves, B=V^T-staged 32x16; 4 d-tiles)
    #pragma unroll
    for (int nt = 0; nt < 4; ++nt) {
      #pragma unroll
      for (int kh = 0; kh < 2; ++kh) {
        BF16x16 ta, tb;
        #pragma unroll
        for (int j = 0; j < 8; ++j) {
          const int kp = (j < 4) ? (hl*4 + j) : (8 + hl*4 + (j - 4));
          ta.u[j] = *(const unsigned int*)&Pl[wave][l15][(kh*16 + kp)*2];
          tb.u[j] = *(const unsigned int*)&Vt[nt*16 + l15][(kh*16 + kp)*2];
        }
        o[nt] = wmma_bf16(ta.v, tb.v, o[nt]);
      }
    }
    __syncthreads();   // K/V tile overwritten next iteration
  }

  // Normalize and write (B*S, E) bf16 attention output.
  const int b = bh / H_CONST, h = bh % H_CONST;
  #pragma unroll
  for (int i = 0; i < 8; ++i) {
    const int qr = q0 + wave*16 + i + 8*hl;
    const float inv = 1.0f / l_i[i];
    #pragma unroll
    for (int nt = 0; nt < 4; ++nt) {
      const int dd = nt*16 + l15;
      attn[((size_t)(b*S_CONST + qr))*E_CONST + h*DHEAD + dd] = f2bf(o[nt][i] * inv);
    }
  }
}

extern "C" void kernel_launch(void* const* d_in, const int* in_sizes, int n_in,
                              void* d_out, int out_size, void* d_ws, size_t ws_size,
                              hipStream_t stream) {
  (void)in_sizes; (void)n_in; (void)out_size; (void)ws_size;
  const float* x     = (const float*)d_in[0];
  const float* w_in  = (const float*)d_in[1];
  const float* b_in  = (const float*)d_in[2];
  const float* w_out = (const float*)d_in[3];
  const float* b_out = (const float*)d_in[4];
  float* out = (float*)d_out;

  // Workspace: Qb | Kb | Vb | attn, each B*H*S*64 bf16 = 8 MB (32 MB total).
  const size_t qkv_elems = (size_t)2 * H_CONST * S_CONST * DHEAD;
  unsigned short* Qb = (unsigned short*)d_ws;
  unsigned short* Kb = Qb + qkv_elems;
  unsigned short* Vb = Kb + qkv_elems;
  unsigned short* At = Vb + qkv_elems;

  const int M = 2 * S_CONST;                                   // 4096 rows

  dim3 g1(M/128, (3*E_CONST)/128);                             // 32 x 24
  gemm_bf16_wmma<float, 0><<<g1, 256, 0, stream>>>(
      x, w_in, b_in, Qb, Kb, Vb, nullptr, M, 3*E_CONST, E_CONST);

  flash_attn_wmma<<<2*H_CONST*(S_CONST/128), 256, 0, stream>>>(Qb, Kb, Vb, At);

  dim3 g2(M/128, E_CONST/128);                                 // 32 x 8
  gemm_bf16_wmma<unsigned short, 1><<<g2, 256, 0, stream>>>(
      At, w_out, b_out, nullptr, nullptr, nullptr, out, M, E_CONST, E_CONST);
}
